// SyncBidirMultiheadAttentionMechanism_16338055594644
// MI455X (gfx1250) — matrix operand order repaired and checked by reading, not compiled, not run on hardware
//
#include <hip/hip_runtime.h>
#include <math.h>

// ---------------------------------------------------------------------------
// SyncBidirMultiheadAttentionMechanism for MI455X (gfx1250, wave32, WMMA)
// B=2, S=1024, KD=AD=512, H=8, DK=64
// Key insight: reference slices cvs[:, :S] only -> only ctx(aw_fwd_h, vf)
// feeds the cv outputs. value_bwd is dead. All 4 attention prob tensors are
// outputs (256MB total -> memory bound, ~11us floor at 23.3 TB/s).
// All GEMMs use v_wmma_f32_16x16x32_bf16 (bf16 operands, fp32 accum).
// ---------------------------------------------------------------------------

#define B_  2
#define S_  1024
#define KD_ 512
#define AD_ 512
#define H_  8
#define DK_ 64
#define NEG_INF_F (-3.402823466e38f)

typedef __attribute__((ext_vector_type(16))) __bf16 v16bf;
typedef __attribute__((ext_vector_type(8)))  float  v8f;

// K-packing for 16-bit A operand (16x32, row per lane):
// VGPR i<4 -> K = 2i,2i+1 ; i>=4 -> K = 16+2(i-4) ; hi half-wave adds +8
__device__ __forceinline__ int ka_off(int i, int hi) {
  return ((i < 4) ? (2 * i) : (16 + 2 * (i - 4))) + (hi ? 8 : 0);
}
// K-packing for 16-bit B operand (32x16, column per lane): same as A:
// VGPR i<4 -> K = 2i,2i+1 ; i>=4 -> K = 16+2(i-4) ; hi half-wave adds +8
__device__ __forceinline__ int kb_off(int i, int hi) {
  return ((i < 4) ? (2 * i) : (16 + 2 * (i - 4))) + (hi ? 8 : 0);
}

__device__ __forceinline__ v8f wmma_bf16(v16bf a, v16bf b, v8f c) {
  return __builtin_amdgcn_wmma_f32_16x16x32_bf16(
      /*neg_a=*/false, a, /*neg_b=*/false, b,
      /*c_mod=*/(short)0, c, /*reuse_a=*/false, /*reuse_b=*/false);
}

// Load a 16-wide K-pair operand from a row-major fp32 matrix (row = per-lane).
__device__ __forceinline__ v16bf load_rowpair_f32(const float* __restrict__ base,
                                                  int rowstride, int row, int k0,
                                                  int lane, bool a_layout) {
  int hi = lane >> 4;
  const float* p = base + (size_t)row * rowstride + k0;
  v16bf v;
#pragma unroll
  for (int i = 0; i < 8; ++i) {
    int kb = a_layout ? ka_off(i, hi) : kb_off(i, hi);
    v[2 * i]     = (__bf16)p[kb];
    v[2 * i + 1] = (__bf16)p[kb + 1];
  }
  return v;
}

// Same but bf16 source.
__device__ __forceinline__ v16bf load_rowpair_bf16(const __bf16* __restrict__ base,
                                                   int rowstride, int row, int k0,
                                                   int lane, bool a_layout) {
  int hi = lane >> 4;
  const __bf16* p = base + (size_t)row * rowstride + k0;
  v16bf v;
#pragma unroll
  for (int i = 0; i < 8; ++i) {
    int kb = a_layout ? ka_off(i, hi) : kb_off(i, hi);
    v[2 * i]     = p[kb];
    v[2 * i + 1] = p[kb + 1];
  }
  return v;
}

// B operand where B[k][n] is row-major [K x N] (for P@V): strided rows.
__device__ __forceinline__ v16bf load_b_kn_bf16(const __bf16* __restrict__ base,
                                                int rowstride, int k0, int n, int hi) {
  v16bf v;
#pragma unroll
  for (int i = 0; i < 8; ++i) {
    int kb = kb_off(i, hi);
    v[2 * i]     = base[(size_t)(k0 + kb) * rowstride + n];
    v[2 * i + 1] = base[(size_t)(k0 + kb + 1) * rowstride + n];
  }
  return v;
}

// ---------------------------------------------------------------------------
// Projection: Y = X @ W^T + bias ; X [2048,512] f32, W [512,512] f32.
// Output stored head-split as bf16 [B,H,S,DK].
// grid (N/16=32, M/16=128), block 32 (one wave per 16x16 tile).
// ---------------------------------------------------------------------------
__global__ __launch_bounds__(32) void proj_heads_kernel(
    const float* __restrict__ X, const float* __restrict__ W,
    const float* __restrict__ bias, __bf16* __restrict__ Y) {
  int lane = threadIdx.x & 31;
  int n0 = blockIdx.x * 16;
  int m0 = blockIdx.y * 16;
  int arow = m0 + (lane & 15);
  int brow = n0 + (lane & 15);
  v8f c = {};
  for (int k0 = 0; k0 < KD_; k0 += 32) {
    v16bf a = load_rowpair_f32(X, KD_, arow, k0, lane, true);
    v16bf b = load_rowpair_f32(W, KD_, brow, k0, lane, false);
    c = wmma_bf16(a, b, c);
  }
  int hi = lane >> 4;
  int col = lane & 15;
  int n = n0 + col;
  int h = n >> 6;   // DK = 64
  int d = n & 63;
  float bv = bias[n];
#pragma unroll
  for (int j = 0; j < 8; ++j) {
    int m = m0 + j + hi * 8;
    int bb = m >> 10;        // S = 1024
    int s = m & 1023;
    Y[(((size_t)bb * H_ + h) * S_ + s) * DK_ + d] = (__bf16)(c[j] + bv);
  }
}

// ---------------------------------------------------------------------------
// Fused attention: e = QK^T / sqrt(DK), mask, softmax, write probs (fp32).
// grid (S/16=64 q-tiles, B*H=16), block 256 (8 waves).
// 16-row x 1024-col logit strip lives in 64KB dynamic LDS.
// ---------------------------------------------------------------------------
__global__ __launch_bounds__(256) void attn_softmax_kernel(
    const __bf16* __restrict__ Q, const __bf16* __restrict__ K,
    const int* __restrict__ mask, float* __restrict__ AW) {
  extern __shared__ float sE[];   // [16][1024]
  int tid  = threadIdx.x;
  int lane = tid & 31;
  int wave = tid >> 5;
  int bh = blockIdx.y;            // b*H + h
  int b  = bh >> 3;
  int q0 = blockIdx.x * 16;
  const __bf16* qp = Q + (size_t)bh * S_ * DK_;
  const __bf16* kp = K + (size_t)bh * S_ * DK_;
  int arow = q0 + (lane & 15);
  int hi  = lane >> 4;
  int col = lane & 15;

  // Q tile operands (16x64 over DK), reused across all 64 key tiles.
  v16bf a0 = load_rowpair_bf16(qp, DK_, arow, 0,  lane, true);
  v16bf a1 = load_rowpair_bf16(qp, DK_, arow, 32, lane, true);

  for (int kt = wave; kt < S_ / 16; kt += 8) {
    int n0 = kt * 16;
    int brow = n0 + col;
    v16bf b0 = load_rowpair_bf16(kp, DK_, brow, 0,  lane, false);
    v16bf b1 = load_rowpair_bf16(kp, DK_, brow, 32, lane, false);
    v8f c = {};
    c = wmma_bf16(a0, b0, c);
    c = wmma_bf16(a1, b1, c);
#pragma unroll
    for (int j = 0; j < 8; ++j) {
      int m = j + hi * 8;                 // tile-local row
      int kcol = n0 + col;
      float e = c[j] * 0.125f;            // 1/sqrt(64)
      int mk = mask[((size_t)b * S_ + (q0 + m)) * S_ + kcol];
      if (mk == 0) e = NEG_INF_F;
      sE[m * S_ + kcol] = e;
    }
  }
  __syncthreads();

  // Softmax: 16 threads per row; thread = (r, sub); 64 strided elems each.
  int r = tid >> 4;
  int sub = tid & 15;
  float* rowp = sE + r * S_;
  float mx = NEG_INF_F;
#pragma unroll 8
  for (int u = 0; u < 64; ++u) mx = fmaxf(mx, rowp[sub + u * 16]);
  mx = fmaxf(mx, __shfl_xor(mx, 1));
  mx = fmaxf(mx, __shfl_xor(mx, 2));
  mx = fmaxf(mx, __shfl_xor(mx, 4));
  mx = fmaxf(mx, __shfl_xor(mx, 8));   // stays inside 16-lane half-wave

  float sum = 0.f;
#pragma unroll 8
  for (int u = 0; u < 64; ++u) {
    float v = __expf(rowp[sub + u * 16] - mx);
    rowp[sub + u * 16] = v;
    sum += v;
  }
  sum += __shfl_xor(sum, 1);
  sum += __shfl_xor(sum, 2);
  sum += __shfl_xor(sum, 4);
  sum += __shfl_xor(sum, 8);
  float inv = 1.0f / sum;

  float* ow = AW + ((size_t)bh * S_ + (q0 + r)) * S_;
#pragma unroll 8
  for (int u = 0; u < 64; ++u) ow[sub + u * 16] = rowp[sub + u * 16] * inv;
}

// ---------------------------------------------------------------------------
// ctx0 = aw_fwd_h @ vf ; per (b,h): [1024x1024] @ [1024x64] -> [1024x64].
// Output layout 'bqhd' flattened: ctx0[(b*S+q)*AD + h*DK + n].
// grid (64, 4, 16), block 32.
// ---------------------------------------------------------------------------
__global__ __launch_bounds__(32) void ctx_kernel(
    const float* __restrict__ AW, const __bf16* __restrict__ V,
    float* __restrict__ CTX) {
  int lane = threadIdx.x & 31;
  int q0 = blockIdx.x * 16;
  int n0 = blockIdx.y * 16;
  int bh = blockIdx.z;
  int b = bh >> 3, h = bh & 7;
  const float*  ap = AW + (size_t)bh * S_ * S_;
  const __bf16* vp = V  + (size_t)bh * S_ * DK_;
  int arow = q0 + (lane & 15);
  int hi = lane >> 4;
  int col = lane & 15;
  int n = n0 + col;
  v8f c = {};
  for (int k0 = 0; k0 < S_; k0 += 32) {
    v16bf a = load_rowpair_f32(ap, S_, arow, k0, lane, true);
    v16bf bm = load_b_kn_bf16(vp, DK_, k0, n, hi);
    c = wmma_bf16(a, bm, c);
  }
#pragma unroll
  for (int j = 0; j < 8; ++j) {
    int q = q0 + j + hi * 8;
    CTX[((size_t)b * S_ + q) * AD_ + h * DK_ + n] = c[j];
  }
}

// ---------------------------------------------------------------------------
// P = ctx0 @ Wo^T + bo ; [2048,512] @ [512,512]^T. grid (32,128), block 32.
// ---------------------------------------------------------------------------
__global__ __launch_bounds__(32) void outproj_kernel(
    const float* __restrict__ CTX, const float* __restrict__ Wo,
    const float* __restrict__ bo, float* __restrict__ P) {
  int lane = threadIdx.x & 31;
  int n0 = blockIdx.x * 16;
  int m0 = blockIdx.y * 16;
  int arow = m0 + (lane & 15);
  int brow = n0 + (lane & 15);
  v8f c = {};
  for (int k0 = 0; k0 < AD_; k0 += 32) {
    v16bf a = load_rowpair_f32(CTX, AD_, arow, k0, lane, true);
    v16bf b = load_rowpair_f32(Wo, AD_, brow, k0, lane, false);
    c = wmma_bf16(a, b, c);
  }
  int hi = lane >> 4, col = lane & 15;
  float bv = bo[n0 + col];
#pragma unroll
  for (int j = 0; j < 8; ++j) {
    int m = m0 + j + hi * 8;
    P[(size_t)m * KD_ + n0 + col] = c[j] + bv;
  }
}

// ---------------------------------------------------------------------------
// cv_fwd = P[:, 0:256] + 0.1*tanh(P[:, 256:512])
// cv_bwd = P[:, 512:768] + 0.1*tanh(P[:, 768:1024])   (rows per batch)
// ---------------------------------------------------------------------------
__global__ __launch_bounds__(256) void combine_kernel(
    const float* __restrict__ P, float* __restrict__ cvf, float* __restrict__ cvb) {
  int idx = blockIdx.x * blockDim.x + threadIdx.x;
  const int total = B_ * 256 * KD_;
  if (idx >= total) return;
  int b = idx / (256 * KD_);
  int rem = idx - b * (256 * KD_);
  int i = rem / KD_;
  int n = rem - i * KD_;
  const float* Pb = P + (size_t)b * S_ * KD_;
  cvf[idx] = Pb[(size_t)i * KD_ + n]         + 0.1f * tanhf(Pb[(size_t)(256 + i) * KD_ + n]);
  cvb[idx] = Pb[(size_t)(512 + i) * KD_ + n] + 0.1f * tanhf(Pb[(size_t)(768 + i) * KD_ + n]);
}

// ---------------------------------------------------------------------------
extern "C" void kernel_launch(void* const* d_in, const int* in_sizes, int n_in,
                              void* d_out, int out_size, void* d_ws, size_t ws_size,
                              hipStream_t stream) {
  (void)in_sizes; (void)n_in; (void)out_size; (void)ws_size;
  const float* key_fwd   = (const float*)d_in[0];
  const float* value_fwd = (const float*)d_in[1];
  const float* query_fwd = (const float*)d_in[2];
  const float* key_bwd   = (const float*)d_in[3];
  /* value_bwd d_in[4] is provably unused by the reference outputs */
  const float* query_bwd = (const float*)d_in[5];
  const int*   mask      = (const int*)d_in[6];
  const float* Wk = (const float*)d_in[7];  const float* bk = (const float*)d_in[8];
  const float* Wv = (const float*)d_in[9];  const float* bv = (const float*)d_in[10];
  const float* Wq = (const float*)d_in[11]; const float* bq = (const float*)d_in[12];
  const float* Wo = (const float*)d_in[13]; const float* bo = (const float*)d_in[14];

  float* out = (float*)d_out;
  char*  ws  = (char*)d_ws;

  const size_t PROJ_ELTS = (size_t)B_ * S_ * AD_;          // 1,048,576
  __bf16* qf = (__bf16*)(ws + 0 * PROJ_ELTS * 2);
  __bf16* kf = (__bf16*)(ws + 1 * PROJ_ELTS * 2);
  __bf16* qb = (__bf16*)(ws + 2 * PROJ_ELTS * 2);
  __bf16* kb = (__bf16*)(ws + 3 * PROJ_ELTS * 2);
  __bf16* vf = (__bf16*)(ws + 4 * PROJ_ELTS * 2);
  float*  ctx0 = (float*)(ws + 5 * PROJ_ELTS * 2);                       // 4MB
  float*  P    = (float*)(ws + 5 * PROJ_ELTS * 2 + PROJ_ELTS * 4);      // 4MB

  // d_out layout: cv_fwd, cv_bwd, aw_fwd_h, aw_fwd_f, aw_bwd_h, aw_bwd_f
  const size_t CV_SZ = (size_t)B_ * 256 * KD_;             // 262,144
  const size_t AW_SZ = (size_t)B_ * H_ * S_ * S_;          // 16,777,216
  float* cv_fwd   = out;
  float* cv_bwd   = out + CV_SZ;
  float* aw_fwd_h = out + 2 * CV_SZ;
  float* aw_fwd_f = aw_fwd_h + AW_SZ;
  float* aw_bwd_h = aw_fwd_f + AW_SZ;
  float* aw_bwd_f = aw_bwd_h + AW_SZ;

  // 1) Projections (5 needed).
  dim3 pg(AD_ / 16, (B_ * S_) / 16);                       // (32, 128)
  proj_heads_kernel<<<pg, 32, 0, stream>>>(key_fwd,   Wk, bk, kf);
  proj_heads_kernel<<<pg, 32, 0, stream>>>(value_fwd, Wv, bv, vf);
  proj_heads_kernel<<<pg, 32, 0, stream>>>(query_fwd, Wq, bq, qf);
  proj_heads_kernel<<<pg, 32, 0, stream>>>(key_bwd,   Wk, bk, kb);
  proj_heads_kernel<<<pg, 32, 0, stream>>>(query_bwd, Wq, bq, qb);

  // 2) Four fused attention softmaxes -> d_out.
  dim3 ag(S_ / 16, B_ * H_);                                // (64, 16)
  size_t smem = (size_t)16 * S_ * sizeof(float);            // 64KB
  attn_softmax_kernel<<<ag, 256, smem, stream>>>(qf, kf, mask, aw_fwd_h);
  attn_softmax_kernel<<<ag, 256, smem, stream>>>(qf, kb, mask, aw_fwd_f);
  attn_softmax_kernel<<<ag, 256, smem, stream>>>(qb, kb, mask, aw_bwd_h);
  attn_softmax_kernel<<<ag, 256, smem, stream>>>(qb, kf, mask, aw_bwd_f);

  // 3) ctx0 = aw_fwd_h @ vf (only context the outputs need).
  dim3 cg(S_ / 16, DK_ / 16, B_ * H_);                      // (64, 4, 16)
  ctx_kernel<<<cg, 32, 0, stream>>>(aw_fwd_h, vf, ctx0);

  // 4) Output projection.
  dim3 og(KD_ / 16, (B_ * S_) / 16);                        // (32, 128)
  outproj_kernel<<<og, 32, 0, stream>>>(ctx0, Wo, bo, P);

  // 5) tanh combine -> cv_fwd / cv_bwd.
  int total = B_ * 256 * KD_;
  combine_kernel<<<(total + 255) / 256, 256, 0, stream>>>(P, cv_fwd, cv_bwd);
}
